// SC2EdgeClassifier_60292750901790
// MI455X (gfx1250) — matrix operand order repaired and checked by reading, not compiled, not run on hardware
//
#include <hip/hip_runtime.h>

typedef __attribute__((ext_vector_type(16))) __bf16 v16bf;
typedef __attribute__((ext_vector_type(8)))  float  v8f;

union ABFrag { v16bf v; uint4 q[2]; };

#define CDIV(a, b) (((a) + (b) - 1) / (b))

// ---------------------------------------------------------------- utilities

__global__ void k_cvt_bf16(const float* __restrict__ in, __bf16* __restrict__ out, int n) {
    int i = blockIdx.x * blockDim.x + threadIdx.x;
    if (i < n) out[i] = (__bf16)in[i];
}

// W [K,N] f32 row-major -> Wt [N,Kpad] bf16 (zero padded for k >= K)
__global__ void k_transpose_pad(const float* __restrict__ W, __bf16* __restrict__ Wt,
                                int K, int N, int Kpad) {
    int idx = blockIdx.x * blockDim.x + threadIdx.x;
    if (idx >= N * Kpad) return;
    int n = idx / Kpad, k = idx - n * Kpad;
    Wt[idx] = (k < K) ? (__bf16)W[(size_t)k * N + n] : (__bf16)0.0f;
}

// ---------------------------------------------------------------- WMMA GEMM
// C[M,ldc] = act(A[M,Kpad](bf16) @ Wt[N,Kpad]^T(bf16) + bias)
// 4 waves per block; each wave owns one 16-row tile and a full 64-column strip:
// one A fragment is reused across 4 WMMAs per k-step (4 independent accumulators).
// grid = (ceil(Mtiles/4), N/64). act: 0=none, 1=ELU, 2=ReLU. Out f32 (Cf) or bf16 (Cb).
__global__ void k_gemm_wmma(const __bf16* __restrict__ A, const __bf16* __restrict__ Wt,
                            const float* __restrict__ bias,
                            float* __restrict__ Cf, __bf16* __restrict__ Cb,
                            int Mtiles, int Kpad, int ldc, int act) {
    const int lane  = threadIdx.x & 31;
    const int w     = threadIdx.x >> 5;
    const int mtile = blockIdx.x * 4 + w;
    if (mtile >= Mtiles) return;            // wave-uniform guard (EXEC stays full)
    const int m0 = mtile * 16;
    const int n0 = blockIdx.y * 64;

    // A fragment (16x32 bf16): lane<16 -> row=lane,    K = k0..k0+15
    //                          lane>=16 -> row=lane-16, K = k0+16..k0+31
    const __bf16* ap = A  + (size_t)(m0 + (lane & 15)) * Kpad + ((lane >> 4) << 4);
    // B fragment (32x16 bf16): lane<16 -> col=lane,     K = k0..k0+15
    //                          lane>=16 -> col=lane-16, K = k0+16..k0+31
    const __bf16* bp = Wt + (size_t)(n0 + (lane & 15)) * Kpad + ((lane >> 4) << 4);
    const size_t bs  = (size_t)16 * Kpad;   // stride between 16-column subtiles

    v8f acc0 = {}, acc1 = {}, acc2 = {}, acc3 = {};
    for (int k0 = 0; k0 < Kpad; k0 += 32) {
        ABFrag a, b0, b1, b2, b3;
        a.q[0]  = *(const uint4*)(ap + k0);
        a.q[1]  = *(const uint4*)(ap + k0 + 8);
        b0.q[0] = *(const uint4*)(bp + k0);
        b0.q[1] = *(const uint4*)(bp + k0 + 8);
        b1.q[0] = *(const uint4*)(bp + bs + k0);
        b1.q[1] = *(const uint4*)(bp + bs + k0 + 8);
        b2.q[0] = *(const uint4*)(bp + 2 * bs + k0);
        b2.q[1] = *(const uint4*)(bp + 2 * bs + k0 + 8);
        b3.q[0] = *(const uint4*)(bp + 3 * bs + k0);
        b3.q[1] = *(const uint4*)(bp + 3 * bs + k0 + 8);
        acc0 = __builtin_amdgcn_wmma_f32_16x16x32_bf16(false, a.v, false, b0.v,
                                                       (short)0, acc0, false, false);
        acc1 = __builtin_amdgcn_wmma_f32_16x16x32_bf16(false, a.v, false, b1.v,
                                                       (short)0, acc1, false, false);
        acc2 = __builtin_amdgcn_wmma_f32_16x16x32_bf16(false, a.v, false, b2.v,
                                                       (short)0, acc2, false, false);
        acc3 = __builtin_amdgcn_wmma_f32_16x16x32_bf16(false, a.v, false, b3.v,
                                                       (short)0, acc3, false, false);
    }

    const int col   = lane & 15;
    const int rbase = m0 + ((lane >> 4) << 3);  // rows r / r+8 split across half-waves
    v8f accs[4] = {acc0, acc1, acc2, acc3};
    #pragma unroll
    for (int j = 0; j < 4; ++j) {
        const int nc  = n0 + j * 16 + col;
        const float bv = bias ? bias[nc] : 0.0f;
        #pragma unroll
        for (int r = 0; r < 8; ++r) {
            float v = accs[j][r] + bv;
            if (act == 1)      v = v > 0.0f ? v : (__expf(v) - 1.0f);   // ELU
            else if (act == 2) v = v > 0.0f ? v : 0.0f;                 // ReLU
            size_t o = (size_t)(rbase + r) * ldc + nc;
            if (Cf) Cf[o] = v;
            else    Cb[o] = (__bf16)v;
        }
    }
}

// ---------------------------------------------------------------- edge attention

__device__ __forceinline__ unsigned f2key(float f) {
    unsigned u = __float_as_uint(f);
    return (u & 0x80000000u) ? ~u : (u | 0x80000000u);
}
__device__ __forceinline__ float key2f(unsigned k) {
    unsigned u = (k & 0x80000000u) ? (k & 0x7FFFFFFFu) : ~k;
    return __uint_as_float(u);
}

// thread = (edge, head). logit = sum_c att[h,c]*leakyrelu(xl[s,h,c]+xr[d,h,c])
__global__ void k_edge_logits(const float* __restrict__ xl, const float* __restrict__ xr,
                              const int* __restrict__ src, const int* __restrict__ dst,
                              const float* __restrict__ att,
                              float* __restrict__ logits, unsigned* __restrict__ mkey,
                              int Ereal, int Etot) {
    int t = blockIdx.x * blockDim.x + threadIdx.x;
    if (t >= Etot * 4) return;
    int e = t >> 2, h = t & 3;
    int s = (e < Ereal) ? src[e] : (e - Ereal);   // self-loops appended
    int d = (e < Ereal) ? dst[e] : (e - Ereal);
    const float4* pl = (const float4*)(xl + (size_t)s * 128 + h * 32);
    const float4* pr = (const float4*)(xr + (size_t)d * 128 + h * 32);
    const float4* pa = (const float4*)(att + h * 32);
    float acc = 0.0f;
    #pragma unroll
    for (int i = 0; i < 8; ++i) {
        float4 a = pl[i], b = pr[i], wv = pa[i];
        float vx = a.x + b.x, vy = a.y + b.y, vz = a.z + b.z, vw = a.w + b.w;
        vx = vx > 0.0f ? vx : 0.2f * vx;
        vy = vy > 0.0f ? vy : 0.2f * vy;
        vz = vz > 0.0f ? vz : 0.2f * vz;
        vw = vw > 0.0f ? vw : 0.2f * vw;
        acc += wv.x * vx + wv.y * vy + wv.z * vz + wv.w * vw;
    }
    logits[t] = acc;
    atomicMax(mkey + (size_t)d * 4 + h, f2key(acc));
}

// thread = (edge, head). ex = exp(logit - max); denom[dst,h] += ex
__global__ void k_edge_expsum(const int* __restrict__ src, const int* __restrict__ dst,
                              float* __restrict__ logits, const unsigned* __restrict__ mkey,
                              float* __restrict__ denom, int Ereal, int Etot) {
    int t = blockIdx.x * blockDim.x + threadIdx.x;
    if (t >= Etot * 4) return;
    int e = t >> 2, h = t & 3;
    int d = (e < Ereal) ? dst[e] : (e - Ereal);
    float m  = key2f(mkey[(size_t)d * 4 + h]);
    float ex = __expf(logits[t] - m);
    logits[t] = ex;                       // reuse buffer for ex
    atomicAdd(denom + (size_t)d * 4 + h, ex);
}

// thread = (edge, group-of-4-channels). out[dst] += alpha * xl[src]
__global__ void k_edge_aggr(const float* __restrict__ xl,
                            const int* __restrict__ src, const int* __restrict__ dst,
                            const float* __restrict__ ex, const float* __restrict__ denom,
                            float* __restrict__ out, int Ereal, int Etot) {
    int t = blockIdx.x * blockDim.x + threadIdx.x;
    if (t >= Etot * 32) return;
    int e = t >> 5, g = t & 31;           // g: 4-channel group, head = g>>3
    int h = g >> 3;
    int s = (e < Ereal) ? src[e] : (e - Ereal);
    int d = (e < Ereal) ? dst[e] : (e - Ereal);
    float alpha = ex[(size_t)e * 4 + h] / denom[(size_t)d * 4 + h];
    float4 v = *(const float4*)(xl + (size_t)s * 128 + g * 4);
    float* o = out + (size_t)d * 128 + g * 4;
    atomicAdd(o + 0, v.x * alpha);
    atomicAdd(o + 1, v.y * alpha);
    atomicAdd(o + 2, v.z * alpha);
    atomicAdd(o + 3, v.w * alpha);
}

// ---------------------------------------------------------------- post / misc

__global__ void k_bias_elu_bf16(const float* __restrict__ in, const float* __restrict__ bias,
                                __bf16* __restrict__ out, int n, int C) {
    int i = blockIdx.x * blockDim.x + threadIdx.x;
    if (i >= n) return;
    int c = i % C;
    float v = in[i] + bias[c];
    out[i] = (__bf16)(v > 0.0f ? v : (__expf(v) - 1.0f));
}

// in [M,4,32] -> out[M,32] = elu(mean_heads + b)
__global__ void k_mean_bias_elu(const float* __restrict__ in, const float* __restrict__ bias,
                                float* __restrict__ out, int M) {
    int i = blockIdx.x * blockDim.x + threadIdx.x;
    if (i >= M * 32) return;
    int n = i >> 5, c = i & 31;
    const float* p = in + (size_t)n * 128 + c;
    float v = 0.25f * (p[0] + p[32] + p[64] + p[96]) + bias[c];
    out[i] = v > 0.0f ? v : (__expf(v) - 1.0f);
}

// feat[q, 0:32]=h2[qs], [32:64]=h2[qd], [64:80]=attr; stride 96, cols 80..95 pre-zeroed
__global__ void k_build_feat(const float* __restrict__ h2, const int* __restrict__ qs,
                             const int* __restrict__ qd, const float* __restrict__ attr,
                             __bf16* __restrict__ feat, int Q) {
    int t = blockIdx.x * blockDim.x + threadIdx.x;
    if (t >= Q * 80) return;
    int q = t / 80, k = t - q * 80;
    float v;
    if (k < 32)      v = h2[(size_t)qs[q] * 32 + k];
    else if (k < 64) v = h2[(size_t)qd[q] * 32 + (k - 32)];
    else             v = attr[(size_t)q * 16 + (k - 64)];
    feat[(size_t)q * 96 + k] = (__bf16)v;
}

// final layer: out[q] = z2[q,:] . W5 + b5   (N=1 -> plain dot product)
__global__ void k_gemv(const __bf16* __restrict__ z2, const float* __restrict__ W5,
                       const float* __restrict__ b5, float* __restrict__ out, int Q) {
    int q = blockIdx.x * blockDim.x + threadIdx.x;
    if (q >= Q) return;
    const __bf16* p = z2 + (size_t)q * 128;
    float acc = b5[0];
    #pragma unroll 4
    for (int k = 0; k < 128; ++k) acc += (float)p[k] * W5[k];
    out[q] = acc;
}

// ---------------------------------------------------------------- host

extern "C" void kernel_launch(void* const* d_in, const int* in_sizes, int n_in,
                              void* d_out, int out_size, void* d_ws, size_t ws_size,
                              hipStream_t stream) {
    constexpr int NN = 50000, EM = 800000, EQ = 200000, ET = EM + NN;
    constexpr int MT_N = NN / 16;      // 3125 m-tiles for node GEMMs
    constexpr int MT_Q = EQ / 16;      // 12500 m-tiles for query GEMMs

    const float* x     = (const float*)d_in[0];
    const int*   msrc  = (const int*)d_in[1];            // [2, EM] flat
    const int*   mdst  = msrc + EM;
    const int*   qs    = (const int*)d_in[2];            // [2, EQ] flat
    const int*   qd    = qs + EQ;
    const float* qattr = (const float*)d_in[3];
    const float* Wl1 = (const float*)d_in[4];  const float* bl1 = (const float*)d_in[5];
    const float* Wr1 = (const float*)d_in[6];  const float* br1 = (const float*)d_in[7];
    const float* att1= (const float*)d_in[8];  const float* b1  = (const float*)d_in[9];
    const float* Wl2 = (const float*)d_in[10]; const float* bl2 = (const float*)d_in[11];
    const float* Wr2 = (const float*)d_in[12]; const float* br2 = (const float*)d_in[13];
    const float* att2= (const float*)d_in[14]; const float* b2  = (const float*)d_in[15];
    const float* W3  = (const float*)d_in[16]; const float* b3  = (const float*)d_in[17];
    const float* W4  = (const float*)d_in[18]; const float* b4  = (const float*)d_in[19];
    const float* W5  = (const float*)d_in[20]; const float* b5  = (const float*)d_in[21];
    float* out = (float*)d_out;

    // ---- workspace carve (256B aligned)
    size_t off = 0;
    auto carve = [&](size_t bytes) -> void* {
        off = (off + 255) & ~(size_t)255;
        void* p = (char*)d_ws + off;
        off += bytes;
        return p;
    };
    __bf16*   xbf   = (__bf16*)carve((size_t)NN * 64 * 2);
    __bf16*   wl1t  = (__bf16*)carve(128 * 64 * 2);
    __bf16*   wr1t  = (__bf16*)carve(128 * 64 * 2);
    __bf16*   wl2t  = (__bf16*)carve(128 * 128 * 2);
    __bf16*   wr2t  = (__bf16*)carve(128 * 128 * 2);
    __bf16*   w3t   = (__bf16*)carve(256 * 96 * 2);
    __bf16*   w4t   = (__bf16*)carve(128 * 256 * 2);
    float*    xl1   = (float*)carve((size_t)NN * 128 * 4);
    float*    xr1   = (float*)carve((size_t)NN * 128 * 4);
    float*    logit = (float*)carve((size_t)ET * 4 * 4);
    unsigned* mkey  = (unsigned*)carve((size_t)NN * 4 * 4);
    float*    denom = (float*)carve((size_t)NN * 4 * 4);
    float*    agg1  = (float*)carve((size_t)NN * 128 * 4);
    __bf16*   h1b   = (__bf16*)carve((size_t)NN * 128 * 2);
    float*    xl2   = (float*)carve((size_t)NN * 128 * 4);
    float*    xr2   = (float*)carve((size_t)NN * 128 * 4);
    float*    agg2  = (float*)carve((size_t)NN * 128 * 4);
    float*    h2    = (float*)carve((size_t)NN * 32 * 4);
    __bf16*   feat  = (__bf16*)carve((size_t)EQ * 96 * 2);
    __bf16*   z1    = (__bf16*)carve((size_t)EQ * 256 * 2);
    __bf16*   z2    = (__bf16*)carve((size_t)EQ * 128 * 2);
    (void)ws_size; (void)n_in; (void)in_sizes; (void)out_size;

    // ---- weight prep
    k_cvt_bf16<<<CDIV(NN * 64, 256), 256, 0, stream>>>(x, xbf, NN * 64);
    k_transpose_pad<<<CDIV(128 * 64, 256), 256, 0, stream>>>(Wl1, wl1t, 64, 128, 64);
    k_transpose_pad<<<CDIV(128 * 64, 256), 256, 0, stream>>>(Wr1, wr1t, 64, 128, 64);
    k_transpose_pad<<<CDIV(128 * 128, 256), 256, 0, stream>>>(Wl2, wl2t, 128, 128, 128);
    k_transpose_pad<<<CDIV(128 * 128, 256), 256, 0, stream>>>(Wr2, wr2t, 128, 128, 128);
    k_transpose_pad<<<CDIV(256 * 96, 256), 256, 0, stream>>>(W3, w3t, 80, 256, 96);
    k_transpose_pad<<<CDIV(128 * 256, 256), 256, 0, stream>>>(W4, w4t, 256, 128, 256);

    // ---- conv1 node linears (WMMA): [50000,64] @ [64,128]
    {
        dim3 g(CDIV(MT_N, 4), 128 / 64);
        k_gemm_wmma<<<g, 128, 0, stream>>>(xbf, wl1t, bl1, xl1, nullptr, MT_N, 64, 128, 0);
        k_gemm_wmma<<<g, 128, 0, stream>>>(xbf, wr1t, br1, xr1, nullptr, MT_N, 64, 128, 0);
    }
    // ---- conv1 segment softmax + aggregation
    hipMemsetAsync(mkey, 0, (size_t)NN * 4 * 4, stream);
    hipMemsetAsync(denom, 0, (size_t)NN * 4 * 4, stream);
    hipMemsetAsync(agg1, 0, (size_t)NN * 128 * 4, stream);
    k_edge_logits<<<CDIV(ET * 4, 256), 256, 0, stream>>>(xl1, xr1, msrc, mdst, att1,
                                                         logit, mkey, EM, ET);
    k_edge_expsum<<<CDIV(ET * 4, 256), 256, 0, stream>>>(msrc, mdst, logit, mkey, denom, EM, ET);
    k_edge_aggr<<<CDIV(ET * 32, 256), 256, 0, stream>>>(xl1, msrc, mdst, logit, denom,
                                                        agg1, EM, ET);
    k_bias_elu_bf16<<<CDIV(NN * 128, 256), 256, 0, stream>>>(agg1, b1, h1b, NN * 128, 128);

    // ---- conv2 node linears (WMMA): [50000,128] @ [128,128]
    {
        dim3 g(CDIV(MT_N, 4), 128 / 64);
        k_gemm_wmma<<<g, 128, 0, stream>>>(h1b, wl2t, bl2, xl2, nullptr, MT_N, 128, 128, 0);
        k_gemm_wmma<<<g, 128, 0, stream>>>(h1b, wr2t, br2, xr2, nullptr, MT_N, 128, 128, 0);
    }
    // ---- conv2 segment softmax + aggregation
    hipMemsetAsync(mkey, 0, (size_t)NN * 4 * 4, stream);
    hipMemsetAsync(denom, 0, (size_t)NN * 4 * 4, stream);
    hipMemsetAsync(agg2, 0, (size_t)NN * 128 * 4, stream);
    k_edge_logits<<<CDIV(ET * 4, 256), 256, 0, stream>>>(xl2, xr2, msrc, mdst, att2,
                                                         logit, mkey, EM, ET);
    k_edge_expsum<<<CDIV(ET * 4, 256), 256, 0, stream>>>(msrc, mdst, logit, mkey, denom, EM, ET);
    k_edge_aggr<<<CDIV(ET * 32, 256), 256, 0, stream>>>(xl2, msrc, mdst, logit, denom,
                                                        agg2, EM, ET);
    k_mean_bias_elu<<<CDIV(NN * 32, 256), 256, 0, stream>>>(agg2, b2, h2, NN);

    // ---- classifier MLP on query edges
    hipMemsetAsync(feat, 0, (size_t)EQ * 96 * 2, stream);    // zero K-padding cols
    k_build_feat<<<CDIV(EQ * 80, 256), 256, 0, stream>>>(h2, qs, qd, qattr, feat, EQ);
    {
        dim3 g1(CDIV(MT_Q, 4), 256 / 64);
        k_gemm_wmma<<<g1, 128, 0, stream>>>(feat, w3t, b3, nullptr, z1, MT_Q, 96, 256, 2);
        dim3 g2(CDIV(MT_Q, 4), 128 / 64);
        k_gemm_wmma<<<g2, 128, 0, stream>>>(z1, w4t, b4, nullptr, z2, MT_Q, 256, 128, 2);
    }
    k_gemv<<<CDIV(EQ, 256), 256, 0, stream>>>(z2, W5, b5, out, EQ);
}
